// PDReconCapsuleRouting_16836271800843
// MI455X (gfx1250) — matrix operand order repaired and checked by reading, not compiled, not run on hardware
//
#include <hip/hip_runtime.h>
#include <stdint.h>

// ---- problem dims ----
#define B_SZ    256
#define NOBJ    20
#define CAPS    128
#define DEC_FEAT (NOBJ*CAPS)   // 2560
#define H1      512
#define H2      1024
#define D_IMG   12288
#define ROWS    (B_SZ*NOBJ)    // 5120
#define RITERS  3

typedef __bf16 v16bf __attribute__((ext_vector_type(16)));
typedef float  v8f   __attribute__((ext_vector_type(8)));

union FragAB { v16bf v; uint4 q[2]; };

__device__ __forceinline__ unsigned short f2bf(float f) {
    unsigned u = __float_as_uint(f);
    u += 0x7FFFu + ((u >> 16) & 1u);   // round-to-nearest-even
    return (unsigned short)(u >> 16);
}
__device__ __forceinline__ float sigmoidf(float v) { return 1.0f / (1.0f + __expf(-v)); }

__device__ __forceinline__ void wait_async0() {
#if __has_builtin(__builtin_amdgcn_s_wait_asynccnt)
    __builtin_amdgcn_s_wait_asynccnt(0);
#else
    asm volatile("s_wait_asynccnt 0x0" ::: "memory");
#endif
}

// async-DMA one 128x32 bf16 tile into LDS: 512 x 16B chunks, 256 threads x 2.
// GLOBAL_LOAD_ASYNC_TO_LDS_B128 (GV mode): per-lane LDS offset VGPR + 64-bit
// global address; tracked by ASYNCcnt, no VGPR data round trip.
__device__ __forceinline__ void stage_async(const unsigned short* __restrict__ P, long rowStart,
                                            int ld, int kb, int t, unsigned short* __restrict__ S) {
    const int r = t >> 2, k8 = (t & 3) * 8;
    const unsigned short* g0 = P + (rowStart + r) * (long)ld + kb + k8;
    const unsigned short* g1 = P + (rowStart + r + 64) * (long)ld + kb + k8;
    unsigned s0 = (unsigned)(size_t)(S + r * 40 + k8);          // low 32 bits = LDS offset
    unsigned s1 = (unsigned)(size_t)(S + (r + 64) * 40 + k8);
    asm volatile("global_load_async_to_lds_b128 %0, %1, off" :: "v"(s0), "v"(g0) : "memory");
    asm volatile("global_load_async_to_lds_b128 %0, %1, off" :: "v"(s1), "v"(g1) : "memory");
}

// ---------------------------------------------------------------------------
// bf16 WMMA GEMM: out = act(A[M,K] @ B[K,N] + bias), with B supplied TRANSPOSED
// as Bt[N][K] (leading dim ldbt). Block: 256 threads = 8 waves (4M x 2N);
// block tile 128(M) x 128(N); wave tile 32(M) x 64(N) = 2x4 WMMA frags;
// K step 32; LDS double-buffered, filled by async global->LDS DMA.
// MODE 0: relu -> bf16 store      (hidden layers)
// MODE 1: sigmoid -> f32 store    (final reconstruction)
// MODE 2: sigmoid -> SSE vs x, per-row half-wave reduce, atomicAdd -> rscore
// blockIdx.z slices A/Bt/out via element offsets aZ/bZ/oZ (layer-1 per-capsule).
// ---------------------------------------------------------------------------
template<int MODE>
__global__ __launch_bounds__(256) void gemm_bf16(
    const unsigned short* __restrict__ A, int lda, long aZ,
    const unsigned short* __restrict__ Bt, int ldbt, long bZ,
    const float* __restrict__ bias,
    void* __restrict__ outp, int ldo, long oZ,
    int K,
    const float* __restrict__ x, int xdiv, int ldx,
    float* __restrict__ rscore)
{
    __shared__ __align__(16) unsigned short As[2][128 * 40];  // [buf][row][k], stride 40 halves
    __shared__ __align__(16) unsigned short Bs[2][128 * 40];  // [buf][col][k], stride 40 halves

    const int t    = threadIdx.x;
    const int lane = t & 31;
    const int wave = t >> 5;
    const int wm   = wave >> 1;          // 0..3 : rows [wm*32, +32)
    const int wn   = wave & 1;           // 0..1 : cols [wn*64, +64)
    const int z    = blockIdx.z;
    const long rowBase = (long)blockIdx.y * 128;
    const long colBase = (long)blockIdx.x * 128;

    A  += (long)z * aZ;
    Bt += (long)z * bZ;

    v8f acc[2][4];
    {
        const v8f zacc = {0.f, 0.f, 0.f, 0.f, 0.f, 0.f, 0.f, 0.f};
        #pragma unroll
        for (int tm = 0; tm < 2; ++tm)
            #pragma unroll
            for (int tn = 0; tn < 4; ++tn) acc[tm][tn] = zacc;
    }

    // prologue: DMA K-step 0 into buffer 0
    stage_async(A,  rowBase, lda,  0, t, As[0]);
    stage_async(Bt, colBase, ldbt, 0, t, Bs[0]);
    wait_async0();
    __syncthreads();

    const int h = lane >> 4, l15 = lane & 15;
    int buf = 0;
    for (int kb = 0; kb < K; kb += 32) {
        const bool hasNext = (kb + 32) < K;
        if (hasNext) {      // kick off DMA for next K-step; overlaps with WMMA below
            stage_async(A,  rowBase, lda,  kb + 32, t, As[buf ^ 1]);
            stage_async(Bt, colBase, ldbt, kb + 32, t, Bs[buf ^ 1]);
        }
        if (kb + 64 < K) {  // L2 prefetch two steps ahead (global_prefetch_b8)
            __builtin_prefetch(A  + (rowBase + (t >> 2)) * (long)lda  + kb + 64, 0, 3);
            __builtin_prefetch(Bt + (colBase + (t >> 2)) * (long)ldbt + kb + 64, 0, 3);
        }

        // ---- fragments per CDNA5 16-bit WMMA VGPR layouts ----
        const unsigned short* as = As[buf];
        const unsigned short* bs = Bs[buf];
        FragAB a[2], b[4];
        #pragma unroll
        for (int tm = 0; tm < 2; ++tm) {
            int mrow = wm * 32 + tm * 16 + l15;
            // lane-half h holds K in [8h,8h+8) and [16+8h,16+8h+8)
            a[tm].q[0] = *(const uint4*)(as + mrow * 40 + 8 * h);
            a[tm].q[1] = *(const uint4*)(as + mrow * 40 + 16 + 8 * h);
        }
        #pragma unroll
        for (int tn = 0; tn < 4; ++tn) {
            int col = wn * 64 + tn * 16 + l15;
            // lane-half h holds K in [8h,8h+8) and [16+8h,16+8h+8) for its column
            b[tn].q[0] = *(const uint4*)(bs + col * 40 + 8 * h);
            b[tn].q[1] = *(const uint4*)(bs + col * 40 + 16 + 8 * h);
        }
        #pragma unroll
        for (int tm = 0; tm < 2; ++tm)
            #pragma unroll
            for (int tn = 0; tn < 4; ++tn)
                acc[tm][tn] = __builtin_amdgcn_wmma_f32_16x16x32_bf16(
                    false, a[tm].v, false, b[tn].v, (short)0, acc[tm][tn], false, false);

        if (hasNext) wait_async0();   // DMA for next buffer must land before barrier
        __syncthreads();
        buf ^= 1;
    }

    // ---- epilogue ----
    float bcol[4];
    #pragma unroll
    for (int tn = 0; tn < 4; ++tn) bcol[tn] = bias[colBase + wn * 64 + tn * 16 + l15];

    if (MODE == 2) {
        float sse[2][8];
        #pragma unroll
        for (int tm = 0; tm < 2; ++tm)
            #pragma unroll
            for (int r = 0; r < 8; ++r) sse[tm][r] = 0.f;
        #pragma unroll
        for (int tm = 0; tm < 2; ++tm)
            #pragma unroll
            for (int r = 0; r < 8; ++r) {
                long row = rowBase + wm * 32 + tm * 16 + r + 8 * h;
                long xro = (row / xdiv) * (long)ldx;
                #pragma unroll
                for (int tn = 0; tn < 4; ++tn) {
                    long col = colBase + wn * 64 + tn * 16 + l15;
                    float v = sigmoidf(acc[tm][tn][r] + bcol[tn]);
                    float e = v - x[xro + col];
                    sse[tm][r] += e * e;
                }
            }
        // half-wave (16-lane) reduce: lanes 0-15 hold row r, lanes 16-31 row r+8
        #pragma unroll
        for (int tm = 0; tm < 2; ++tm)
            #pragma unroll
            for (int r = 0; r < 8; ++r) {
                float v = sse[tm][r];
                v += __shfl_xor(v, 1, 32);
                v += __shfl_xor(v, 2, 32);
                v += __shfl_xor(v, 4, 32);
                v += __shfl_xor(v, 8, 32);
                if (l15 == 0) {
                    long row = rowBase + wm * 32 + tm * 16 + r + 8 * h;
                    atomicAdd(&rscore[row], v);
                }
            }
    } else if (MODE == 1) {
        float* out = (float*)outp;
        #pragma unroll
        for (int tm = 0; tm < 2; ++tm)
            #pragma unroll
            for (int tn = 0; tn < 4; ++tn)
                #pragma unroll
                for (int r = 0; r < 8; ++r) {
                    long row = rowBase + wm * 32 + tm * 16 + r + 8 * h;
                    long col = colBase + wn * 64 + tn * 16 + l15;
                    out[row * (long)ldo + col] = sigmoidf(acc[tm][tn][r] + bcol[tn]);
                }
    } else {
        unsigned short* out = (unsigned short*)outp + (long)z * oZ;
        #pragma unroll
        for (int tm = 0; tm < 2; ++tm)
            #pragma unroll
            for (int tn = 0; tn < 4; ++tn)
                #pragma unroll
                for (int r = 0; r < 8; ++r) {
                    long row = rowBase + wm * 32 + tm * 16 + r + 8 * h;
                    long col = colBase + wn * 64 + tn * 16 + l15;
                    float v = fmaxf(acc[tm][tn][r] + bcol[tn], 0.f);
                    out[row * (long)ldo + col] = f2bf(v);
                }
    }
}

// ---------------------------------------------------------------------------
// transpose + convert: in f32 [K,N] -> out bf16 [N,K]; LDS-tiled, fully coalesced
__global__ __launch_bounds__(256) void cvt_transpose_f32_bf16(
    const float* __restrict__ in, unsigned short* __restrict__ out, int K, int N)
{
    __shared__ float tile[32][33];
    const int nb = blockIdx.x * 32, kb = blockIdx.y * 32;
    const int tx = threadIdx.x & 31;
    const int ty = threadIdx.x >> 5;
    #pragma unroll
    for (int r = ty; r < 32; r += 8)
        tile[r][tx] = in[(long)(kb + r) * N + nb + tx];
    __syncthreads();
    #pragma unroll
    for (int r = ty; r < 32; r += 8)
        out[(long)(nb + r) * K + kb + tx] = f2bf(tile[tx][r]);
}

__global__ void init_caps(const float* __restrict__ in, float* __restrict__ cf,
                          unsigned short* __restrict__ cb, long n) {
    long i = (long)blockIdx.x * blockDim.x + threadIdx.x;
    long stride = (long)gridDim.x * blockDim.x;
    for (; i < n; i += stride) { float v = in[i]; cf[i] = v; cb[i] = f2bf(v); }
}

__global__ void zero_f32(float* __restrict__ p, int n) {
    int i = blockIdx.x * blockDim.x + threadIdx.x;
    if (i < n) p[i] = 0.f;
}

// per-sample routing coefficients + squash; 1 block per sample, 128 threads
__global__ __launch_bounds__(128) void routing_update(
    const float* __restrict__ rscore, float* __restrict__ capsF, unsigned short* __restrict__ capsBf)
{
    const int b = blockIdx.x;
    const int t = threadIdx.x;
    __shared__ float sc[NOBJ];
    __shared__ float red[4];
    if (t < NOBJ) sc[t] = rscore[b * NOBJ + t];   // s = SSE (positive); rscore_ref = -s
    __syncthreads();
    float smin = sc[0], smax = sc[0];
    #pragma unroll
    for (int j = 1; j < NOBJ; ++j) { smin = fminf(smin, sc[j]); smax = fmaxf(smax, sc[j]); }
    float denom = fmaxf(smax - smin, 1e-6f);

    for (int j = 0; j < NOBJ; ++j) {
        float coef = fmaxf((smax - sc[j]) / denom, 0.5f);   // == (r-mn)/(mx-mn) with r=-s
        long idx = ((long)b * NOBJ + j) * CAPS + t;
        float v = capsF[idx] * coef;
        float s2 = v * v;
        s2 += __shfl_xor(s2, 1, 32);
        s2 += __shfl_xor(s2, 2, 32);
        s2 += __shfl_xor(s2, 4, 32);
        s2 += __shfl_xor(s2, 8, 32);
        s2 += __shfl_xor(s2, 16, 32);
        if ((t & 31) == 0) red[t >> 5] = s2;
        __syncthreads();
        float tot = red[0] + red[1] + red[2] + red[3];
        float nrm = sqrtf(tot);
        float scale = (nrm * nrm) / (1.f + nrm * nrm) / (nrm + 1e-8f);
        float o = v * scale;
        capsF[idx] = o;
        capsBf[idx] = f2bf(o);
        __syncthreads();
    }
}

// ---------------------------------------------------------------------------
extern "C" void kernel_launch(void* const* d_in, const int* in_sizes, int n_in,
                              void* d_out, int out_size, void* d_ws, size_t ws_size,
                              hipStream_t stream) {
    const float* x  = (const float*)d_in[0];
    const float* oc = (const float*)d_in[1];
    const float* W1 = (const float*)d_in[2];
    const float* b1 = (const float*)d_in[3];
    const float* W2 = (const float*)d_in[4];
    const float* b2 = (const float*)d_in[5];
    const float* W3 = (const float*)d_in[6];
    const float* b3 = (const float*)d_in[7];

    uint8_t* base = (uint8_t*)d_ws;
    size_t off = 0;
    auto carve = [&](size_t bytes) -> void* {
        void* p = base + off;
        off = (off + bytes + 255) & ~(size_t)255;
        return p;
    };
    // transposed bf16 weights: Wt[N][K]
    unsigned short* w1t    = (unsigned short*)carve((size_t)DEC_FEAT * H1 * 2);
    unsigned short* w2t    = (unsigned short*)carve((size_t)H1 * H2 * 2);
    unsigned short* w3t    = (unsigned short*)carve((size_t)H2 * D_IMG * 2);
    float*          capsF  = (float*)carve((size_t)B_SZ * NOBJ * CAPS * 4);
    unsigned short* capsBf = (unsigned short*)carve((size_t)B_SZ * NOBJ * CAPS * 2);
    unsigned short* h1     = (unsigned short*)carve((size_t)ROWS * H1 * 2);
    unsigned short* h2     = (unsigned short*)carve((size_t)ROWS * H2 * 2);
    float*          rscore = (float*)carve((size_t)ROWS * 4);

    // weights -> bf16 transposed (once per launch; amortized over 3 routing iters)
    cvt_transpose_f32_bf16<<<dim3(H1 / 32, DEC_FEAT / 32), 256, 0, stream>>>(W1, w1t, DEC_FEAT, H1);
    cvt_transpose_f32_bf16<<<dim3(H2 / 32, H1 / 32), 256, 0, stream>>>(W2, w2t, H1, H2);
    cvt_transpose_f32_bf16<<<dim3(D_IMG / 32, H2 / 32), 256, 0, stream>>>(W3, w3t, H2, D_IMG);
    init_caps<<<1024, 256, 0, stream>>>(oc, capsF, capsBf, (long)B_SZ * NOBJ * CAPS);

    for (int it = 0; it < RITERS; ++it) {
        zero_f32<<<ROWS / 256, 256, 0, stream>>>(rscore, ROWS);

        // L1 (masked): per capsule j, [256,128] @ W1[j*128:(j+1)*128, :512]
        // Bt slice = w1t[:, j*128 : j*128+128]  (k-offset j*128, ldbt = DEC_FEAT)
        gemm_bf16<0><<<dim3(H1 / 128, B_SZ / 128, NOBJ), 256, 0, stream>>>(
            capsBf, DEC_FEAT, (long)CAPS,
            w1t, DEC_FEAT, (long)CAPS,
            b1, h1, NOBJ * H1, (long)H1,
            CAPS, nullptr, 1, 0, nullptr);

        // L2: [5120,512] @ [512,1024]
        gemm_bf16<0><<<dim3(H2 / 128, ROWS / 128, 1), 256, 0, stream>>>(
            h1, H1, 0L, w2t, H1, 0L, b2, h2, H2, 0L,
            H1, nullptr, 1, 0, nullptr);

        // L3 fused with routing score: [5120,1024] @ [1024,12288], sigmoid+SSE -> rscore
        gemm_bf16<2><<<dim3(D_IMG / 128, ROWS / 128, 1), 256, 0, stream>>>(
            h2, H2, 0L, w3t, H2, 0L, b3, nullptr, 0, 0L,
            H2, x, NOBJ, D_IMG, rscore);

        routing_update<<<B_SZ, 128, 0, stream>>>(rscore, capsF, capsBf);
    }

    // final decode from routed capsules: capsBf is exactly [256, 2560] row-major
    gemm_bf16<0><<<dim3(H1 / 128, B_SZ / 128, 1), 256, 0, stream>>>(
        capsBf, DEC_FEAT, 0L, w1t, DEC_FEAT, 0L, b1, h1, H1, 0L,
        DEC_FEAT, nullptr, 1, 0, nullptr);
    gemm_bf16<0><<<dim3(H2 / 128, B_SZ / 128, 1), 256, 0, stream>>>(
        h1, H1, 0L, w2t, H1, 0L, b2, h2, H2, 0L,
        H1, nullptr, 1, 0, nullptr);
    gemm_bf16<1><<<dim3(D_IMG / 128, B_SZ / 128, 1), 256, 0, stream>>>(
        h2, H2, 0L, w3t, H2, 0L, b3, d_out, D_IMG, 0L,
        H2, nullptr, 1, 0, nullptr);
}
